// PS_Attention_76768245448786
// MI455X (gfx1250) — matrix operand order repaired and checked by reading, not compiled, not run on hardware
//
#include <hip/hip_runtime.h>

typedef __attribute__((ext_vector_type(16))) _Float16 v16h;
typedef __attribute__((ext_vector_type(8)))  float    v8f;

#define HW 3136   // 56*56

// ---------- helpers ----------
__device__ __forceinline__ v16h ld16(const _Float16* p) {
  v16h r;
  reinterpret_cast<float4*>(&r)[0] = reinterpret_cast<const float4*>(p)[0];
  reinterpret_cast<float4*>(&r)[1] = reinterpret_cast<const float4*>(p)[1];
  return r;
}

__device__ __forceinline__ v8f wmma16(v16h a, v16h b, v8f c) {
  // D = A(16x32 f16) x B(32x16 f16) + C(16x16 f32)
  return __builtin_amdgcn_wmma_f32_16x16x32_f16(false, a, false, b, (short)0, c,
                                                false, false);
}

// ---------- 1) depthwise 3x3 + BN -> f16, stored in WMMA B-operand layout ----------
// Bact layout: [i(3)][b(4)][kchunk(8)][ntile(196)][lane(32)][elem(16)] f16
// B-matrix element (lane, j): K-index = (lane>>4)*16 + j, column n = lane&15
__global__ void dwbn_kernel(const float* __restrict__ x, const float* __restrict__ dw,
                            const float* __restrict__ gamma, const float* __restrict__ beta,
                            const float* __restrict__ mean, const float* __restrict__ var,
                            _Float16* __restrict__ Bact) {
  int tid = blockIdx.x * 256 + threadIdx.x;        // 3*4*256*3136 threads
  int s = tid % HW;
  int q = tid / HW;
  int c = q & 255;
  int b = (q >> 8) & 3;
  int i = q >> 10;
  int h = s / 56, w = s - h * 56;
  const float* xp = x + (b * 256 + c) * HW;
  const float* wp = dw + (i * 256 + c) * 9;
  float acc = 0.f;
#pragma unroll
  for (int dy = 0; dy < 3; ++dy) {
    int hh = h + dy - 1;
    if (hh < 0 || hh > 55) continue;
#pragma unroll
    for (int dx = 0; dx < 3; ++dx) {
      int ww = w + dx - 1;
      if (ww < 0 || ww > 55) continue;
      acc += xp[hh * 56 + ww] * wp[dy * 3 + dx];
    }
  }
  float inv = gamma[i * 256 + c] * rsqrtf(var[i * 256 + c] + 1e-5f);
  float o = acc * inv + (beta[i * 256 + c] - mean[i * 256 + c] * inv);
  int lane = ((c >> 4) & 1) * 16 + (s & 15);
  int j = c & 15;
  Bact[((((i * 4 + b) * 8 + (c >> 5)) * 196 + (s >> 4)) * 32 + lane) * 16 + j] = (_Float16)o;
}

// ---------- 2) pack 1x1-conv weights into WMMA A-operand layout ----------
// A-matrix element (lane, j): row m = lane&15, K-index = (lane>>4)*16 + j
__global__ void pack_w_kernel(const float* __restrict__ qkv_pw, const float* __restrict__ proj_w,
                              _Float16* __restrict__ Wq, _Float16* __restrict__ Wp) {
  int tid = blockIdx.x * 256 + threadIdx.x;        // 4*16*8*512 = 262144 threads
  int j = tid & 15;
  int lane = (tid >> 4) & 31;
  int kc = (tid >> 9) & 7;
  int mt = (tid >> 12) & 15;
  int ii = tid >> 16;                              // 0..2 = qkv, 3 = proj
  int co = mt * 16 + (lane & 15);
  int lg = lane >> 4;
  int kl = lg * 16 + j;
  int ci = kc * 32 + kl;
  if (ii < 3) {
    Wq[(((ii * 16 + mt) * 8 + kc) * 32 + lane) * 16 + j] =
        (_Float16)qkv_pw[(ii * 256 + co) * 256 + ci];
  } else {
    Wp[((mt * 8 + kc) * 32 + lane) * 16 + j] = (_Float16)proj_w[co * 256 + ci];
  }
}

// ---------- 3) pointwise qkv GEMM (WMMA) with attention-layout scatter epilogue ----------
// Qb: [axis2][win32][head8][qtile25][lane32][elem16]  (B-operand: lane=16*(d>>3)+token, elem=d&7, elems8-15 zero)
// Kb: [axis2][win32][head8][ktile26][lane32][elem16]  (A-operand: m=token, K=d padded to 32)
// Vb: [axis2][win32][head8][kchunk13][lane32][elem16] (A-operand: m=d, K=token-in-chunk)
__global__ void gemm_qkv_kernel(const _Float16* __restrict__ Wq, const _Float16* __restrict__ Bact,
                                _Float16* __restrict__ Qb, _Float16* __restrict__ Kb,
                                _Float16* __restrict__ Vb) {
  int wid = blockIdx.x * 8 + (threadIdx.x >> 5);   // 3*4*16*196 waves
  int lane = threadIdx.x & 31;
  int nt = wid % 196;
  int q2 = wid / 196;
  int mt = q2 & 15;
  int rb = q2 >> 4;
  int b = rb & 3;
  int i = rb >> 2;
  v8f acc = {0.f, 0.f, 0.f, 0.f, 0.f, 0.f, 0.f, 0.f};
#pragma unroll
  for (int kc = 0; kc < 8; ++kc) {
    v16h a  = ld16(Wq + ((i * 16 + mt) * 8 + kc) * 512 + lane * 16);
    v16h bm = ld16(Bact + (((i * 4 + b) * 8 + kc) * 196 + nt) * 512 + lane * 16);
    acc = wmma16(a, bm, acc);
  }
  int n = lane & 15, lg = lane >> 4;
  int s = nt * 16 + n;
  int h = s / 56, w = s - h * 56;
#pragma unroll
  for (int r = 0; r < 8; ++r) {
    int co = mt * 16 + lg * 8 + r;
    float val = acc[r];
    int axis = co >> 7;
    int ch = co & 127;
    int head = ch >> 4;
    int d = ch & 15;
    int win, t;
    if (axis == 0) { win = (b << 3) | (h & 7); t = (h >> 3) * 56 + w; }
    else           { win = (b << 3) | (w & 7); t = h * 7 + (w >> 3); }
    int base = (axis * 32 + win) * 8 + head;
    if (i == 0) {
      Qb[(base * 25 + (t >> 4)) * 512 + ((d >> 3) * 16 + (t & 15)) * 16 + (d & 7)] =
          (_Float16)(val * 0.17677669529663687f);   // fold scale = 1/sqrt(32)
    } else if (i == 1) {
      Kb[(base * 26 + (t >> 4)) * 512 + ((d >> 3) * 16 + (t & 15)) * 16 + (d & 7)] =
          (_Float16)val;
    } else {
      int kl = t & 31;
      Vb[(base * 13 + (t >> 5)) * 512 + (((kl >> 3) & 1) * 16 + d) * 16 +
         (((kl & 16) >> 1) | (kl & 7))] = (_Float16)val;
    }
  }
}

// ---------- 4) flash-style windowed attention (one q-tile per wave) ----------
// Yact: proj-GEMM B-operand layout, same scheme as Bact (per batch)
__global__ void attn_kernel(const _Float16* __restrict__ Qb, const _Float16* __restrict__ Kb,
                            const _Float16* __restrict__ Vb, _Float16* __restrict__ Yact) {
  int wid = blockIdx.x * 8 + (threadIdx.x >> 5);   // 2*32*8*25 = 12800 waves
  int lane = threadIdx.x & 31;
  int lg = lane >> 4;
  int qt = wid % 25;
  int u = wid / 25;
  int head = u & 7;
  int u2 = u >> 3;
  int win = u2 & 31;
  int axis = u2 >> 5;
  int base = (axis * 32 + win) * 8 + head;

  v16h qv = ld16(Qb + (base * 25 + qt) * 512 + lane * 16);
  const v8f zero = {0.f, 0.f, 0.f, 0.f, 0.f, 0.f, 0.f, 0.f};
  v8f o = zero;
  float rmax = -1e30f, rsum = 0.f;

  for (int ck = 0; ck < 13; ++ck) {
    v16h ka0 = ld16(Kb + (base * 26 + 2 * ck) * 512 + lane * 16);
    v16h ka1 = ld16(Kb + (base * 26 + 2 * ck + 1) * 512 + lane * 16);
    v8f s0 = wmma16(ka0, qv, zero);   // S^T tile: rows = k tokens 0..15, cols = q
    v8f s1 = wmma16(ka1, qv, zero);   // S^T tile: rows = k tokens 16..31
    float e0[8], e1[8];
    float cm = -1e30f;
#pragma unroll
    for (int r = 0; r < 8; ++r) {
      float a0 = s0[r], a1 = s1[r];
      if (ck == 12) {                 // tokens >= 392 are padding
        if (lg) a0 = -1e30f;
        a1 = -1e30f;
      }
      e0[r] = a0; e1[r] = a1;
      cm = fmaxf(cm, fmaxf(a0, a1));
    }
    cm = fmaxf(cm, __shfl_xor(cm, 16, 32));   // lane pair holds same q
    float nm = fmaxf(rmax, cm);
    float corr = __expf(rmax - nm);
    rmax = nm;
    float loc = 0.f;
#pragma unroll
    for (int r = 0; r < 8; ++r) {
      e0[r] = __expf(e0[r] - nm);
      e1[r] = __expf(e1[r] - nm);
      loc += e0[r] + e1[r];
    }
    rsum = rsum * corr + loc;
    // re-pack exp(S^T) into the P^T B-operand: lane needs K = lg*8 + j (j<8), 16 + lg*8 + (j-8) (j>=8)
    float f0[8], f1[8];
#pragma unroll
    for (int r = 0; r < 8; ++r) {
      f0[r] = __shfl_xor(e0[r], 16, 32);
      f1[r] = __shfl_xor(e1[r], 16, 32);
    }
    v16h pB;
#pragma unroll
    for (int r = 0; r < 8; ++r) {
      pB[r]     = (_Float16)e0[r];
      pB[r + 8] = (_Float16)e1[r];
    }
    v16h va = ld16(Vb + (base * 13 + ck) * 512 + lane * 16);
#pragma unroll
    for (int r = 0; r < 8; ++r) o[r] *= corr;
    o = wmma16(va, pB, o);            // O[d,q] += V x P^T
  }
  float tot = rsum + __shfl_xor(rsum, 16, 32);
  float inv = 1.0f / tot;
  int n = lane & 15;
  int t = qt * 16 + n;
  if (t < 392) {
    int b = win >> 3, g = win & 7;
    int s;
    if (axis == 0) { int p = t / 56; s = (p * 8 + g) * 56 + (t - p * 56); }
    else           { int hh = t / 7; s = hh * 56 + ((t - hh * 7) * 8 + g); }
#pragma unroll
    for (int r = 0; r < 8; ++r) {
      int c = axis * 128 + head * 16 + lg * 8 + r;
      Yact[(((b * 8 + (c >> 5)) * 196 + (s >> 4)) * 32 +
            (((c >> 4) & 1) * 16 + (s & 15))) * 16 + (c & 15)] = (_Float16)(o[r] * inv);
    }
  }
}

// ---------- 5) projection GEMM (WMMA) + bias -> fp32 output ----------
__global__ void gemm_proj_kernel(const _Float16* __restrict__ Wp, const _Float16* __restrict__ Yact,
                                 const float* __restrict__ projb, float* __restrict__ out) {
  int wid = blockIdx.x * 8 + (threadIdx.x >> 5);   // 4*16*196 waves
  int lane = threadIdx.x & 31;
  int nt = wid % 196;
  int q2 = wid / 196;
  int mt = q2 & 15;
  int b = q2 >> 4;
  v8f acc = {0.f, 0.f, 0.f, 0.f, 0.f, 0.f, 0.f, 0.f};
#pragma unroll
  for (int kc = 0; kc < 8; ++kc) {
    v16h a  = ld16(Wp + (mt * 8 + kc) * 512 + lane * 16);
    v16h bm = ld16(Yact + ((b * 8 + kc) * 196 + nt) * 512 + lane * 16);
    acc = wmma16(a, bm, acc);
  }
  int s = nt * 16 + (lane & 15);
  int lg = lane >> 4;
#pragma unroll
  for (int r = 0; r < 8; ++r) {
    int co = mt * 16 + lg * 8 + r;
    out[(b * 256 + co) * HW + s] = acc[r] + projb[co];
  }
}

// ---------- workspace layout (bytes) ----------
#define OFF_BACT  0u                 // 3*4*8*196*1024      = 19267584
#define OFF_WQKV  19267584u          // 3*16*8*1024         =   393216
#define OFF_WPROJ 19660800u          // 16*8*1024           =   131072
#define OFF_Q     19791872u          // 2*32*8*25*1024      = 13107200
#define OFF_K     32899072u          // 2*32*8*26*1024      = 13631488
#define OFF_V     46530560u          // 2*32*8*13*1024      =  6815744
#define OFF_Y     53346304u          // 4*8*196*1024        =  6422528
// total = 59768832 bytes

extern "C" void kernel_launch(void* const* d_in, const int* in_sizes, int n_in,
                              void* d_out, int out_size, void* d_ws, size_t ws_size,
                              hipStream_t stream) {
  (void)in_sizes; (void)n_in; (void)out_size; (void)ws_size;
  const float* x     = (const float*)d_in[0];
  const float* dw    = (const float*)d_in[1];
  const float* gamma = (const float*)d_in[2];
  const float* beta  = (const float*)d_in[3];
  const float* mean  = (const float*)d_in[4];
  const float* var   = (const float*)d_in[5];
  const float* pw    = (const float*)d_in[6];
  const float* pjw   = (const float*)d_in[7];
  const float* pjb   = (const float*)d_in[8];

  char* ws = (char*)d_ws;
  _Float16* Bact = (_Float16*)(ws + OFF_BACT);
  _Float16* Wq   = (_Float16*)(ws + OFF_WQKV);
  _Float16* Wp   = (_Float16*)(ws + OFF_WPROJ);
  _Float16* Qb   = (_Float16*)(ws + OFF_Q);
  _Float16* Kb   = (_Float16*)(ws + OFF_K);
  _Float16* Vb   = (_Float16*)(ws + OFF_V);
  _Float16* Yact = (_Float16*)(ws + OFF_Y);

  // zero Q/K/V padding slots (d=16..31, tokens>=392) every launch — deterministic
  (void)hipMemsetAsync(ws + OFF_Q, 0, 33554432u, stream);

  pack_w_kernel   <<<1024,  256, 0, stream>>>(pw, pjw, Wq, Wp);
  dwbn_kernel     <<<37632, 256, 0, stream>>>(x, dw, gamma, beta, mean, var, Bact);
  gemm_qkv_kernel <<<4704,  256, 0, stream>>>(Wq, Bact, Qb, Kb, Vb);
  attn_kernel     <<<1600,  256, 0, stream>>>(Qb, Kb, Vb, Yact);
  gemm_proj_kernel<<<1568,  256, 0, stream>>>(Wp, Yact, pjb, (float*)d_out);
}
